// DeformableTransformerDecoderLayer_13486197309486
// MI455X (gfx1250) — compile-verified, matrix operand order checked by reading, not run on hardware
//
#include <hip/hip_runtime.h>
#include <hip/hip_bf16.h>

// ---------------------------------------------------------------------------
// Deformable DETR decoder layer for MI455X (gfx1250, wave32, WMMA).
// All GEMMs: f32 -> bf16 packed-convert into LDS (double-buffered, BK=64),
// v_wmma_f32_16x16x32_bf16, f32 accumulate.
// ---------------------------------------------------------------------------

typedef __attribute__((ext_vector_type(16))) __bf16        v16bf;
typedef __attribute__((ext_vector_type(8)))  float         v8f;
typedef __attribute__((ext_vector_type(4)))  float         v4f;
typedef __attribute__((ext_vector_type(2)))  unsigned int  v2u;

// Problem constants (fixed by the reference module).
#define BB   4
#define NQ   900
#define DM   256
#define HH   8
#define DH   32
#define LL   4
#define PP   4
#define DFF  1024
#define NV   21260
#define MROW 3600          // B*NQ

// packed f32x2 -> bf16x2 in ONE v_cvt_pk_bf16_f32 (dst.lo=cvt(x), dst.hi=cvt(y))
__device__ __forceinline__ unsigned int pkbf(float x, float y) {
    unsigned int r;
    asm("v_cvt_pk_bf16_f32 %0, %1, %2" : "=v"(r) : "v"(x), "v"(y));
    return r;
}

// ---------------------------------------------------------------------------
// Tiled bf16 WMMA GEMM, double-buffered LDS, branch-free edge handling.
//   TRANS_B = true : Bm is [N][K] row-major (the x @ W.T case)
//   TRANS_B = false: Bm is [K][N] row-major (plain A @ B)
//   ACT: 0 = none, 1 = ReLU
//   OUTMODE: 0 = C[z*cBatch + m*N + n]
//            1 = split-heads store: (B*NQ, 256) -> (B, H, NQ, 32)
//            2 = merge-heads store: batch z=(b*8+h), (900,32) -> (B, NQ, 256)
// Block: 256 threads = 8 waves, tile 64x64, K-stage 64 (2 WMMA k-substeps).
// Each wave computes a 16(M) x 32(N) slab = two 16x16 WMMA accumulators,
// 4 v_wmma per stage between one barrier pair.
// NOTE: requires K % 4 == 0 (true for all call sites: 256, 1024, 32, 900).
// ---------------------------------------------------------------------------
template<bool TRANS_B, int ACT, int OUTMODE>
__global__ void __launch_bounds__(256)
gemm_bf16_wmma(const float* __restrict__ A, const float* __restrict__ Bm,
               const float* __restrict__ bias, float* __restrict__ C,
               int M, int N, int K,
               long long aBatch, long long bBatch, long long cBatch)
{
    // row stride 72 shorts = 144B: 8B-aligned rows, 36*r mod 64 conflict-free
    __shared__ unsigned short As[2][64][72];
    __shared__ unsigned short Bs[2][64][72];   // stored [n][k]

    const int tid  = threadIdx.x;
    const int lane = tid & 31;
    const int wave = tid >> 5;
    const int wm   = wave >> 1;     // 0..3 : wave row in M
    const int wn   = wave & 1;      // 0..1 : wave col in N
    const int half = lane >> 4;
    const int l16  = lane & 15;
    const int m0   = blockIdx.y * 64;
    const int n0   = blockIdx.x * 64;
    const int z    = blockIdx.z;

    A  += (size_t)z * aBatch;
    Bm += (size_t)z * bBatch;

    v8f acc0 = {};
    v8f acc1 = {};

    v4f apf[4];                     // A prefetch: 4 quads/thread (64x64 tile)
    v4f bpf[4];                     // B prefetch (TRANS_B path)
    float bpeA[8], bpeB[8];         // B prefetch (K-major path: k-even/k-odd)
    const v4f zero4 = {};

    // ---- branch-free global fetch (clamped address + select-to-zero) ----
    auto fetchA = [&](int k0) {
        #pragma unroll
        for (int i = 0; i < 4; ++i) {
            int q  = tid + i * 256;            // 0..1023
            int r  = q >> 4, qc = q & 15;      // row 0..63, quad 0..15 (k=qc*4)
            int gm = m0 + r, gk = k0 + qc * 4;
            int sm = gm < M ? gm : M - 1;
            int sk = (gk + 4 <= K) ? gk : 0;
            v4f v = *(const v4f*)(A + (size_t)sm * K + sk);
            if (!(gm < M && gk + 4 <= K)) v = zero4;
            apf[i] = v;
        }
    };
    auto fetchBt = [&](int k0) {               // Bm[N][K]
        #pragma unroll
        for (int i = 0; i < 4; ++i) {
            int q  = tid + i * 256;
            int r  = q >> 4, qc = q & 15;
            int gn = n0 + r, gk = k0 + qc * 4;
            int sn = gn < N ? gn : N - 1;
            int sk = (gk + 4 <= K) ? gk : 0;
            v4f v = *(const v4f*)(Bm + (size_t)sn * K + sk);
            if (!(gn < N && gk + 4 <= K)) v = zero4;
            bpf[i] = v;
        }
    };
    auto fetchBk = [&](int k0) {               // Bm[K][N], coalesced along n
        #pragma unroll
        for (int i = 0; i < 8; ++i) {
            int idx = tid + i * 256;           // 0..2047 (n x kpair)
            int pc = idx >> 6, n = idx & 63;   // kpair 0..31, n 0..63
            int gn = n0 + n, gk = k0 + pc * 2;
            int sn = gn < N ? gn : N - 1;
            int sk0 = gk     < K ? gk     : K - 1;
            int sk1 = gk + 1 < K ? gk + 1 : K - 1;
            float f0 = Bm[(size_t)sk0 * N + sn];
            float f1 = Bm[(size_t)sk1 * N + sn];
            bpeA[i] = (gn < N && gk     < K) ? f0 : 0.0f;
            bpeB[i] = (gn < N && gk + 1 < K) ? f1 : 0.0f;
        }
    };
    // ---- regs -> LDS (packed bf16 convert, b64 stores) ----
    auto stashA = [&](int p) {
        #pragma unroll
        for (int i = 0; i < 4; ++i) {
            int q = tid + i * 256;
            int r = q >> 4, qc = q & 15;
            v2u pk;
            pk.x = pkbf(apf[i].x, apf[i].y);
            pk.y = pkbf(apf[i].z, apf[i].w);
            *(v2u*)&As[p][r][qc * 4] = pk;
        }
    };
    auto stashBt = [&](int p) {
        #pragma unroll
        for (int i = 0; i < 4; ++i) {
            int q = tid + i * 256;
            int r = q >> 4, qc = q & 15;
            v2u pk;
            pk.x = pkbf(bpf[i].x, bpf[i].y);
            pk.y = pkbf(bpf[i].z, bpf[i].w);
            *(v2u*)&Bs[p][r][qc * 4] = pk;
        }
    };
    auto stashBk = [&](int p) {
        #pragma unroll
        for (int i = 0; i < 8; ++i) {
            int idx = tid + i * 256;
            int pc = idx >> 6, n = idx & 63;
            *(unsigned int*)&Bs[p][n][pc * 2] = pkbf(bpeA[i], bpeB[i]);
        }
    };

    const int ksteps = (K + 63) >> 6;

    // pipeline prologue
    fetchA(0);
    if (TRANS_B) fetchBt(0); else fetchBk(0);
    stashA(0);
    if (TRANS_B) stashBt(0); else stashBk(0);
    __syncthreads();

    for (int ks = 0; ks < ksteps; ++ks) {
        const int p = ks & 1;
        if (ks + 1 < ksteps) {                 // global loads fly over the WMMAs
            fetchA((ks + 1) * 64);
            if (TRANS_B) fetchBt((ks + 1) * 64); else fetchBk((ks + 1) * 64);
        }
        #pragma unroll
        for (int sub = 0; sub < 2; ++sub) {    // two K=32 substeps per stage
            // fragments per ISA 7.12.2 16-bit layout
            union { unsigned int u[8]; v16bf v; } af, bf0, bf1;
            #pragma unroll
            for (int j = 0; j < 8; ++j) {
                int kk = sub * 32 + ((j >> 2) << 4) + (half << 3) + ((j & 3) << 1);
                af.u[j]  = *(const unsigned int*)&As[p][wm * 16 + l16][kk];
                bf0.u[j] = *(const unsigned int*)&Bs[p][wn * 32 + l16][kk];
                bf1.u[j] = *(const unsigned int*)&Bs[p][wn * 32 + 16 + l16][kk];
            }
            acc0 = __builtin_amdgcn_wmma_f32_16x16x32_bf16(false, af.v, false, bf0.v,
                                                           (short)0, acc0, false, false);
            acc1 = __builtin_amdgcn_wmma_f32_16x16x32_bf16(false, af.v, false, bf1.v,
                                                           (short)0, acc1, false, false);
        }
        if (ks + 1 < ksteps) {
            __syncthreads();                   // everyone done reading buf 1-p
            stashA(1 - p);
            if (TRANS_B) stashBt(1 - p); else stashBk(1 - p);
            __syncthreads();                   // buf 1-p visible for next iter
        }
    }

    // ---- store (C/D layout: vgpr r, lanes 0-15 -> M=r, lanes 16-31 -> M=r+8) ----
    #pragma unroll
    for (int r = 0; r < 8; ++r) {
        int gm = m0 + wm * 16 + half * 8 + r;
        #pragma unroll
        for (int t = 0; t < 2; ++t) {
            int gn = n0 + wn * 32 + t * 16 + l16;
            if (gm < M && gn < N) {
                float val = (t == 0) ? acc0[r] : acc1[r];
                if (bias) val += bias[gn];
                if (ACT == 1) val = fmaxf(val, 0.0f);
                size_t oi;
                if (OUTMODE == 0) {
                    oi = (size_t)z * cBatch + (size_t)gm * N + gn;
                } else if (OUTMODE == 1) {        // (B*NQ,256) -> (B,H,NQ,32)
                    int b = gm / NQ, q = gm - b * NQ;
                    int h = gn >> 5, d = gn & 31;
                    oi = (((size_t)b * HH + h) * NQ + q) * DH + d;
                } else {                           // z=(b*8+h), (900,32) -> (B,NQ,256)
                    int b = z >> 3, h = z & 7;
                    oi = ((size_t)b * NQ + gm) * DM + h * DH + gn;
                }
                C[oi] = val;
            }
        }
    }
}

// ---------------------------------------------------------------------------
__global__ void add_kernel(const float* __restrict__ a, const float* __restrict__ b,
                           float* __restrict__ c, int n)
{
    int i = blockIdx.x * 256 + threadIdx.x;
    if (i < n) c[i] = a[i] + b[i];
}

// out = LayerNorm(x + y) * g + b ; one block per row, D == 256 == blockDim
__global__ void __launch_bounds__(256)
add_ln_kernel(const float* __restrict__ x, const float* __restrict__ y,
              const float* __restrict__ g, const float* __restrict__ b,
              float* __restrict__ out)
{
    const int t = threadIdx.x;
    const size_t row = blockIdx.x;
    float v = x[row * DM + t] + y[row * DM + t];
    __shared__ float red[256];
    red[t] = v; __syncthreads();
    #pragma unroll
    for (int s = 128; s > 0; s >>= 1) { if (t < s) red[t] += red[t + s]; __syncthreads(); }
    float mean = red[0] * (1.0f / 256.0f);
    __syncthreads();
    float dv = v - mean;
    red[t] = dv * dv; __syncthreads();
    #pragma unroll
    for (int s = 128; s > 0; s >>= 1) { if (t < s) red[t] += red[t + s]; __syncthreads(); }
    float var = red[0] * (1.0f / 256.0f);
    out[row * DM + t] = dv * rsqrtf(var + 1e-5f) * g[t] + b[t];
}

// in-place row softmax with pre-scale; one block per row
__global__ void __launch_bounds__(256)
softmax_scale_kernel(float* __restrict__ x, int cols, float scale)
{
    const size_t row = blockIdx.x;
    float* p = x + row * (size_t)cols;
    const int t = threadIdx.x;
    __shared__ float red[256];
    float mx = -3.4e38f;
    for (int c = t; c < cols; c += 256) mx = fmaxf(mx, p[c] * scale);
    red[t] = mx; __syncthreads();
    #pragma unroll
    for (int s = 128; s > 0; s >>= 1) { if (t < s) red[t] = fmaxf(red[t], red[t + s]); __syncthreads(); }
    mx = red[0]; __syncthreads();
    float sum = 0.0f;
    for (int c = t; c < cols; c += 256) { float e = __expf(p[c] * scale - mx); p[c] = e; sum += e; }
    red[t] = sum; __syncthreads();
    #pragma unroll
    for (int s = 128; s > 0; s >>= 1) { if (t < s) red[t] += red[t + s]; __syncthreads(); }
    float inv = 1.0f / red[0];
    for (int c = t; c < cols; c += 256) p[c] *= inv;
}

// ---------------------------------------------------------------------------
// Multi-scale deformable attention gather (branch-free bilinear corners).
// Grid = B*NQ blocks; 256 threads: h = t>>5, d = t&31.
// off: (B*NQ, H*L*P*2), awl: (B*NQ, H*L*P) logits, v: (B, NV, H, DH)
// ---------------------------------------------------------------------------
__global__ void __launch_bounds__(256)
deform_attn_kernel(const float* __restrict__ refp, const float* __restrict__ off,
                   const float* __restrict__ awl, const float* __restrict__ v,
                   float* __restrict__ out)
{
    const int LVL_H[LL] = {100, 50, 25, 13};
    const int LVL_W[LL] = {160, 80, 40, 20};
    const int LVL_S[LL] = {0, 16000, 20000, 21000};

    const int bq = blockIdx.x;
    const int b  = bq / NQ;
    const int t  = threadIdx.x;
    const int h  = t >> 5;
    const int d  = t & 31;

    // per-head softmax over L*P = 16 attention logits (redundant across d lanes)
    const float* aw = awl + (size_t)bq * (HH * LL * PP) + h * (LL * PP);
    float wv[LL * PP];
    float mx = -3.4e38f;
    #pragma unroll
    for (int i = 0; i < LL * PP; ++i) { wv[i] = aw[i]; mx = fmaxf(mx, wv[i]); }
    float s = 0.0f;
    #pragma unroll
    for (int i = 0; i < LL * PP; ++i) { wv[i] = __expf(wv[i] - mx); s += wv[i]; }
    const float inv = 1.0f / s;

    const float* offp = off + (size_t)bq * (HH * LL * PP * 2) + h * (LL * PP * 2);
    float acc = 0.0f;

    #pragma unroll
    for (int l = 0; l < LL; ++l) {
        const int Hl = LVL_H[l], Wl = LVL_W[l];
        const float rx = refp[((size_t)bq * LL + l) * 2 + 0];
        const float ry = refp[((size_t)bq * LL + l) * 2 + 1];
        const float* vl = v + ((size_t)b * NV + LVL_S[l]) * DM + h * DH + d;
        #pragma unroll
        for (int p = 0; p < PP; ++p) {
            const float ox = offp[(l * PP + p) * 2 + 0];
            const float oy = offp[(l * PP + p) * 2 + 1];
            const float x = (rx + ox / (float)Wl) * (float)Wl - 0.5f;
            const float y = (ry + oy / (float)Hl) * (float)Hl - 0.5f;
            const float x0f = floorf(x), y0f = floorf(y);
            const float wx = x - x0f, wy = y - y0f;
            const int x0 = (int)x0f, y0 = (int)y0f;
            float sum = 0.0f;
            #pragma unroll
            for (int cy = 0; cy < 2; ++cy) {
                #pragma unroll
                for (int cx = 0; cx < 2; ++cx) {
                    int xi = x0 + cx, yi = y0 + cy;
                    int xs = min(max(xi, 0), Wl - 1);
                    int ys = min(max(yi, 0), Hl - 1);
                    float w = (cx ? wx : 1.0f - wx) * (cy ? wy : 1.0f - wy);
                    bool ok = (xi >= 0 && xi < Wl && yi >= 0 && yi < Hl);
                    float val = vl[(size_t)(ys * Wl + xs) * DM];
                    sum += val * (ok ? w : 0.0f);
                }
            }
            acc += wv[l * PP + p] * inv * sum;
        }
    }
    out[(size_t)bq * DM + t] = acc;
}

// ---------------------------------------------------------------------------
// Workspace layout (floats). Peak = 41,126,400 floats ~= 165 MB.
// The big region holds v_proj (87MB) first, then (after v dies) scores (104MB).
// ---------------------------------------------------------------------------
static const size_t F_Q   = 0;
static const size_t F_OFF = F_Q   + 921600;
static const size_t F_AW  = F_OFF + 921600;
static const size_t F_T2A = F_AW  + 460800;
static const size_t F_T2B = F_T2A + 921600;
static const size_t F_X1  = F_T2B + 921600;
static const size_t F_QK  = F_X1  + 921600;
static const size_t F_QH  = F_QK  + 921600;
static const size_t F_KH  = F_QH  + 921600;
static const size_t F_VH  = F_KH  + 921600;
static const size_t F_ATT = F_VH  + 921600;
static const size_t F_X2  = F_ATT + 921600;
static const size_t F_FFH = F_X2  + 921600;
static const size_t F_FF  = F_FFH + 3686400;
static const size_t F_BIG = F_FF  + 921600;   // v_proj, later scores

extern "C" void kernel_launch(void* const* d_in, const int* in_sizes, int n_in,
                              void* d_out, int out_size, void* d_ws, size_t ws_size,
                              hipStream_t stream)
{
    (void)in_sizes; (void)n_in; (void)out_size; (void)ws_size;
    const float* tgt  = (const float*)d_in[0];
    const float* pos  = (const float*)d_in[1];
    const float* refp = (const float*)d_in[2];
    const float* mem  = (const float*)d_in[3];
    const float* Wv   = (const float*)d_in[4];  const float* bv  = (const float*)d_in[5];
    const float* Wo   = (const float*)d_in[6];  const float* bo  = (const float*)d_in[7];
    const float* Wa   = (const float*)d_in[8];  const float* ba  = (const float*)d_in[9];
    const float* Wca  = (const float*)d_in[10]; const float* bca = (const float*)d_in[11];
    const float* g1   = (const float*)d_in[12]; const float* bt1 = (const float*)d_in[13];
    const float* Win  = (const float*)d_in[14]; const float* bin = (const float*)d_in[15];
    const float* Wsa  = (const float*)d_in[16]; const float* bsa = (const float*)d_in[17];
    const float* g2   = (const float*)d_in[18]; const float* bt2 = (const float*)d_in[19];
    const float* W1   = (const float*)d_in[20]; const float* bf1 = (const float*)d_in[21];
    const float* W2   = (const float*)d_in[22]; const float* bf2 = (const float*)d_in[23];
    const float* g3   = (const float*)d_in[24]; const float* bt3 = (const float*)d_in[25];
    float* ws  = (float*)d_ws;
    float* out = (float*)d_out;

    const dim3 blk(256);
    const int MB = 85040;                       // B * NV

    // 1. q = tgt + pos
    add_kernel<<<MROW, blk, 0, stream>>>(tgt, pos, ws + F_Q, MROW * DM);

    // 2. v = memory @ Wv^T + bv  (dominant GEMM: 85040 x 256 x 256)
    gemm_bf16_wmma<true, 0, 0><<<dim3(4, (MB + 63) / 64, 1), blk, 0, stream>>>(
        mem, Wv, bv, ws + F_BIG, MB, DM, DM, 0, 0, 0);

    // 3. sampling offsets: 3600 x 256 x 256
    gemm_bf16_wmma<true, 0, 0><<<dim3(4, 57, 1), blk, 0, stream>>>(
        ws + F_Q, Wo, bo, ws + F_OFF, MROW, 256, DM, 0, 0, 0);

    // 4. attention-weight logits: 3600 x 128 x 256
    gemm_bf16_wmma<true, 0, 0><<<dim3(2, 57, 1), blk, 0, stream>>>(
        ws + F_Q, Wa, ba, ws + F_AW, MROW, 128, DM, 0, 0, 0);

    // 5. deformable bilinear gather
    deform_attn_kernel<<<MROW, blk, 0, stream>>>(refp, ws + F_OFF, ws + F_AW,
                                                 ws + F_BIG, ws + F_T2A);

    // 6. ca output proj
    gemm_bf16_wmma<true, 0, 0><<<dim3(4, 57, 1), blk, 0, stream>>>(
        ws + F_T2A, Wca, bca, ws + F_T2B, MROW, DM, DM, 0, 0, 0);

    // 7. x1 = LN(tgt + tgt2)
    add_ln_kernel<<<MROW, blk, 0, stream>>>(tgt, ws + F_T2B, g1, bt1, ws + F_X1);

    // 8. qk = x1 + pos
    add_kernel<<<MROW, blk, 0, stream>>>(ws + F_X1, pos, ws + F_QK, MROW * DM);

    // 9-11. Q/K/V projections, split-heads store -> (B,H,NQ,32)
    gemm_bf16_wmma<true, 0, 1><<<dim3(4, 57, 1), blk, 0, stream>>>(
        ws + F_QK, Win, bin, ws + F_QH, MROW, DM, DM, 0, 0, 0);
    gemm_bf16_wmma<true, 0, 1><<<dim3(4, 57, 1), blk, 0, stream>>>(
        ws + F_QK, Win + DM * DM, bin + DM, ws + F_KH, MROW, DM, DM, 0, 0, 0);
    gemm_bf16_wmma<true, 0, 1><<<dim3(4, 57, 1), blk, 0, stream>>>(
        ws + F_X1, Win + 2 * DM * DM, bin + 2 * DM, ws + F_VH, MROW, DM, DM, 0, 0, 0);

    // 12. scores = qh @ kh^T, batched over 32 (b,h); v_proj region is dead now
    gemm_bf16_wmma<true, 0, 0><<<dim3(15, 15, BB * HH), blk, 0, stream>>>(
        ws + F_QH, ws + F_KH, nullptr, ws + F_BIG, NQ, NQ, DH,
        (long long)NQ * DH, (long long)NQ * DH, (long long)NQ * NQ);

    // 13. softmax(scores / sqrt(32))
    softmax_scale_kernel<<<BB * HH * NQ, blk, 0, stream>>>(
        ws + F_BIG, NQ, 0.17677669529663687f);

    // 14. attn_out = probs @ vh (A@B, merge-heads store -> (B,NQ,256))
    gemm_bf16_wmma<false, 0, 2><<<dim3(1, 15, BB * HH), blk, 0, stream>>>(
        ws + F_BIG, ws + F_VH, nullptr, ws + F_ATT, NQ, DH, NQ,
        (long long)NQ * NQ, (long long)NQ * DH, 0);

    // 15. self-attn output proj (reuse F_T2B)
    gemm_bf16_wmma<true, 0, 0><<<dim3(4, 57, 1), blk, 0, stream>>>(
        ws + F_ATT, Wsa, bsa, ws + F_T2B, MROW, DM, DM, 0, 0, 0);

    // 16. x2 = LN(x1 + sa)
    add_ln_kernel<<<MROW, blk, 0, stream>>>(ws + F_X1, ws + F_T2B, g2, bt2, ws + F_X2);

    // 17. FFN1 + ReLU: 3600 x 1024 x 256
    gemm_bf16_wmma<true, 1, 0><<<dim3(16, 57, 1), blk, 0, stream>>>(
        ws + F_X2, W1, bf1, ws + F_FFH, MROW, DFF, DM, 0, 0, 0);

    // 18. FFN2: 3600 x 256 x 1024
    gemm_bf16_wmma<true, 0, 0><<<dim3(4, 57, 1), blk, 0, stream>>>(
        ws + F_FFH, W2, bf2, ws + F_FF, MROW, DM, DFF, 0, 0, 0);

    // 19. out = LN(x2 + ff)
    add_ln_kernel<<<MROW, blk, 0, stream>>>(ws + F_X2, ws + F_FF, g3, bt3, out);
}
